// GraphNeuralNetworkRiskManager_14491219657471
// MI455X (gfx1250) — matrix operand order repaired and checked by reading, not compiled, not run on hardware
//
#include <hip/hip_runtime.h>

// ---------------------------------------------------------------------------
// GNN risk manager for MI455X (gfx1250):
//   bf16 WMMA GEMMs with async-to-LDS A-tile staging + hw-atomic scatters.
// ---------------------------------------------------------------------------

#define N_NODES 50000
#define N_EDGES 800000
#define HID     128
#define F_OUT   2
#define LDS_STRIDE 136   // bf16 elems per staged row (128 + 8 pad -> bank rotate)

typedef __bf16 bf16_t;
typedef __attribute__((ext_vector_type(16))) __bf16 v16bf;
typedef __attribute__((ext_vector_type(8)))  __bf16 v8bf;
typedef __attribute__((ext_vector_type(4)))  __bf16 v4bf;
typedef __attribute__((ext_vector_type(8)))  float  v8f;

__device__ __forceinline__ void atomicAddF(float* p, float v) {
  // relaxed agent-scope fp32 add -> global_atomic_add_f32
  __hip_atomic_fetch_add(p, v, __ATOMIC_RELAXED, __HIP_MEMORY_SCOPE_AGENT);
}

// Monotonic float<->uint key so segment-max can use global_atomic_max_u32.
__device__ __forceinline__ unsigned fkey(float f) {
  unsigned u = __float_as_uint(f);
  return (u & 0x80000000u) ? ~u : (u | 0x80000000u);
}
__device__ __forceinline__ float funkey(unsigned k) {
  return __uint_as_float((k & 0x80000000u) ? (k & 0x7fffffffu) : ~k);
}

// ---------------------------------------------------------------------------
// Utility kernels
// ---------------------------------------------------------------------------
__global__ void k_zero4(float* __restrict__ p, long long n4) {
  long long i = (long long)blockIdx.x * blockDim.x + threadIdx.x;
  if (i < n4) { float4 z = {0.f, 0.f, 0.f, 0.f}; *(float4*)(p + i * 4) = z; }
}

__global__ void k_cvt_bf16(const float* __restrict__ in, bf16_t* __restrict__ out,
                           long long n4) {
  long long i = (long long)blockIdx.x * blockDim.x + threadIdx.x;
  if (i >= n4) return;
  float4 v = *(const float4*)(in + i * 4);
  v4bf o = {(__bf16)v.x, (__bf16)v.y, (__bf16)v.z, (__bf16)v.w};
  *(v4bf*)(out + i * 4) = o;
}

// Wt[n][k] = (bf16)W[k][n] for a 128x128 weight
__global__ void k_wtrans(const float* __restrict__ W, bf16_t* __restrict__ Wt) {
  int i = blockIdx.x * blockDim.x + threadIdx.x;
  if (i >= HID * HID) return;
  int n = i >> 7, k = i & 127;
  Wt[i] = (__bf16)W[(size_t)k * HID + n];
}

// ---------------------------------------------------------------------------
// Edge scatter: agg[dst] += feat[src]   (one wave per edge, float4 per lane)
// ---------------------------------------------------------------------------
__global__ void k_scatter_add(const float* __restrict__ feat,
                              const int* __restrict__ src,
                              const int* __restrict__ dst,
                              float* __restrict__ agg, int E) {
  long long tid = (long long)blockIdx.x * blockDim.x + threadIdx.x;
  int e = (int)(tid >> 5);
  if (e >= E) return;
  int c = ((int)tid & 31) * 4;
  int s = src[e], d = dst[e];
  float4 v = *(const float4*)(feat + (size_t)s * HID + c);
  float* p = agg + (size_t)d * HID + c;
  atomicAddF(p + 0, v.x); atomicAddF(p + 1, v.y);
  atomicAddF(p + 2, v.z); atomicAddF(p + 3, v.w);
}

// ---------------------------------------------------------------------------
// Dual bf16 WMMA GEMM: out = act( A1@W1 + (dual ? A2@W2 : 0) + bias )
// One block = one 16-row M tile; 8 waves = 8 N tiles (HID/16).
// A tiles are staged once per block into LDS via global_load_async_to_lds_b128
// (ASYNCcnt), then each wave builds WMMA fragments with ds_load_b128.
// Weights (L2-resident, pre-transposed Wt[n][k]) are read directly from global.
// ---------------------------------------------------------------------------
__global__ __launch_bounds__(256)
void k_gemm_bf16(const bf16_t* __restrict__ A1, const bf16_t* __restrict__ A2,
                 const bf16_t* __restrict__ W1t, const bf16_t* __restrict__ W2t,
                 const float* __restrict__ bias, float* __restrict__ out,
                 int dual, int fuse_relu) {
  __shared__ bf16_t smem[2 * 16 * LDS_STRIDE];      // 8.5 KB of 320 KB WGP LDS
  const int tid  = threadIdx.x;
  const int lane = tid & 31;
  const int tn   = tid >> 5;                        // wave id -> N tile
  const int tm   = blockIdx.x;                      // M tile
  const int g    = lane >> 4;                       // lane half
  const int r    = lane & 15;

  bf16_t* sA1 = smem;
  bf16_t* sA2 = smem + 16 * LDS_STRIDE;

  // ---- async stage: each thread moves one 16B chunk of the 16x128 tile ----
  {
    const int row   = tid >> 4;                     // 0..15
    const int chunk = tid & 15;                     // 16 x 16B = 256B row
    unsigned lds1 = (unsigned)(size_t)(sA1 + row * LDS_STRIDE) + chunk * 16;
    unsigned long long g1 =
        (unsigned long long)(size_t)(A1 + (size_t)(tm * 16 + row) * HID) + chunk * 16;
    asm volatile("global_load_async_to_lds_b128 %0, %1, off"
                 :: "v"(lds1), "v"(g1) : "memory");
    if (dual) {
      unsigned lds2 = (unsigned)(size_t)(sA2 + row * LDS_STRIDE) + chunk * 16;
      unsigned long long g2 =
          (unsigned long long)(size_t)(A2 + (size_t)(tm * 16 + row) * HID) + chunk * 16;
      asm volatile("global_load_async_to_lds_b128 %0, %1, off"
                   :: "v"(lds2), "v"(g2) : "memory");
    }
    asm volatile("s_wait_asynccnt 0x0" ::: "memory");
  }
  __syncthreads();

  const bf16_t* w1row = W1t + (size_t)(tn * 16 + r) * HID;
  const bf16_t* w2row = W2t + (size_t)(tn * 16 + r) * HID;
  const bf16_t* a1row = sA1 + r * LDS_STRIDE;
  const bf16_t* a2row = sA2 + r * LDS_STRIDE;

  v8f c = {};
#pragma unroll
  for (int kk = 0; kk < HID; kk += 32) {
    // A fragment (16x32): K = kk + 16g + e (contiguous, matches B)
    v8bf alo = *(const v8bf*)(a1row + kk + 16 * g);
    v8bf ahi = *(const v8bf*)(a1row + kk + 16 * g + 8);
    v16bf a;
#pragma unroll
    for (int i = 0; i < 8; ++i) { a[i] = alo[i]; a[8 + i] = ahi[i]; }
    // B fragment (32x16): lane col n, K = kk + 16g + e  (contiguous in Wt)
    v16bf b = *(const v16bf*)(w1row + kk + 16 * g);
    c = __builtin_amdgcn_wmma_f32_16x16x32_bf16(false, a, false, b,
                                                (short)0, c, false, false);
    if (dual) {
      v8bf blo = *(const v8bf*)(a2row + kk + 16 * g);
      v8bf bhi = *(const v8bf*)(a2row + kk + 16 * g + 8);
      v16bf a2;
#pragma unroll
      for (int i = 0; i < 8; ++i) { a2[i] = blo[i]; a2[8 + i] = bhi[i]; }
      v16bf b2 = *(const v16bf*)(w2row + kk + 16 * g);
      c = __builtin_amdgcn_wmma_f32_16x16x32_bf16(false, a2, false, b2,
                                                  (short)0, c, false, false);
    }
  }
  const float bv = bias ? bias[tn * 16 + r] : 0.0f;
#pragma unroll
  for (int v = 0; v < 8; ++v) {       // C/D: N=lane%16, M=v+8g
    float val = c[v] + bv;
    if (fuse_relu) val = val > 0.0f ? val : 0.0f;
    out[(size_t)(tm * 16 + v + 8 * g) * HID + tn * 16 + r] = val;
  }
}

// ---------------------------------------------------------------------------
// Attention scores: s_src[i] = ht[i].a_src ; s_dst[i] = ht[i].a_dst
// ---------------------------------------------------------------------------
__global__ void k_attn_scores(const float* __restrict__ ht,
                              const float* __restrict__ a_src,
                              const float* __restrict__ a_dst,
                              float* __restrict__ sS, float* __restrict__ sD,
                              int n) {
  int wave = (int)((blockIdx.x * blockDim.x + threadIdx.x) >> 5);
  int lane = threadIdx.x & 31;
  if (wave >= n) return;
  float4 h  = *(const float4*)(ht + (size_t)wave * HID + lane * 4);
  float4 as = *(const float4*)(a_src + lane * 4);
  float4 ad = *(const float4*)(a_dst + lane * 4);
  float ss = h.x * as.x + h.y * as.y + h.z * as.z + h.w * as.w;
  float sd = h.x * ad.x + h.y * ad.y + h.z * ad.z + h.w * ad.w;
#pragma unroll
  for (int off = 16; off > 0; off >>= 1) {
    ss += __shfl_xor(ss, off, 32);
    sd += __shfl_xor(sd, off, 32);
  }
  if (lane == 0) { sS[wave] = ss; sD[wave] = sd; }
}

__global__ void k_edge_logits(const float* __restrict__ sS,
                              const float* __restrict__ sD,
                              const int* __restrict__ src,
                              const int* __restrict__ dst,
                              float* __restrict__ eV,
                              unsigned* __restrict__ mkey, int E) {
  int e = blockIdx.x * blockDim.x + threadIdx.x;
  if (e >= E) return;
  float v = sS[src[e]] + sD[dst[e]];
  v = v > 0.0f ? v : 0.2f * v;                 // leaky_relu(0.2)
  eV[e] = v;
  atomicMax(&mkey[dst[e]], fkey(v));           // global_atomic_max_u32
}

__global__ void k_edge_exp(const float* __restrict__ eV,
                           const unsigned* __restrict__ mkey,
                           const int* __restrict__ dst,
                           float* __restrict__ eE,
                           float* __restrict__ denom, int E) {
  int e = blockIdx.x * blockDim.x + threadIdx.x;
  if (e >= E) return;
  int d = dst[e];
  float x = __expf(eV[e] - funkey(mkey[d]));
  eE[e] = x;
  atomicAddF(&denom[d], x);
}

__global__ void k_weighted_scatter(const float* __restrict__ eE,
                                   const float* __restrict__ denom,
                                   const int* __restrict__ src,
                                   const int* __restrict__ dst,
                                   const float* __restrict__ ht,
                                   float* __restrict__ of, int E) {
  long long tid = (long long)blockIdx.x * blockDim.x + threadIdx.x;
  int e = (int)(tid >> 5);
  if (e >= E) return;
  int c = ((int)tid & 31) * 4;
  int s = src[e], d = dst[e];
  float alpha = eE[e] / (denom[d] + 1e-16f);
  float4 v = *(const float4*)(ht + (size_t)s * HID + c);
  float* p = of + (size_t)d * HID + c;
  atomicAddF(p + 0, alpha * v.x); atomicAddF(p + 1, alpha * v.y);
  atomicAddF(p + 2, alpha * v.z); atomicAddF(p + 3, alpha * v.w);
}

// Final FC: out[i, 0:2] = of[i] @ Wfc + bfc   (one wave per node)
__global__ void k_final_fc(const float* __restrict__ of,
                           const float* __restrict__ Wfc,
                           const float* __restrict__ bfc,
                           float* __restrict__ out, int n) {
  int wave = (int)((blockIdx.x * blockDim.x + threadIdx.x) >> 5);
  int lane = threadIdx.x & 31;
  if (wave >= n) return;
  float4 h = *(const float4*)(of + (size_t)wave * HID + lane * 4);
  const float* wr = Wfc + lane * 8;            // rows 4l..4l+3 of [128,2]
  float4 w01 = *(const float4*)(wr);
  float4 w23 = *(const float4*)(wr + 4);
  float s0 = h.x * w01.x + h.y * w01.z + h.z * w23.x + h.w * w23.z;
  float s1 = h.x * w01.y + h.y * w01.w + h.z * w23.y + h.w * w23.w;
#pragma unroll
  for (int off = 16; off > 0; off >>= 1) {
    s0 += __shfl_xor(s0, off, 32);
    s1 += __shfl_xor(s1, off, 32);
  }
  if (lane == 0) {
    out[(size_t)wave * 2 + 0] = s0 + bfc[0];
    out[(size_t)wave * 2 + 1] = s1 + bfc[1];
  }
}

// ---------------------------------------------------------------------------
// Host launcher
// ---------------------------------------------------------------------------
extern "C" void kernel_launch(void* const* d_in, const int* in_sizes, int n_in,
                              void* d_out, int out_size, void* d_ws, size_t ws_size,
                              hipStream_t stream) {
  (void)in_sizes; (void)n_in; (void)out_size; (void)ws_size;
  const float* x      = (const float*)d_in[0];
  const int*   eidx   = (const int*)d_in[1];
  const float* W1r    = (const float*)d_in[2];
  const float* W1n    = (const float*)d_in[3];
  const float* b1     = (const float*)d_in[4];
  const float* W2r    = (const float*)d_in[5];
  const float* W2n    = (const float*)d_in[6];
  const float* b2     = (const float*)d_in[7];
  const float* Wa     = (const float*)d_in[8];
  const float* a_src  = (const float*)d_in[9];
  const float* a_dst  = (const float*)d_in[10];
  const float* Wfc    = (const float*)d_in[11];
  const float* bfc    = (const float*)d_in[12];
  float*       out    = (float*)d_out;
  const int* src = eidx;
  const int* dst = eidx + N_EDGES;

  // --- workspace carve-up (256B aligned) ---
  char* base = (char*)d_ws;
  size_t off = 0;
  auto alloc = [&](size_t bytes) -> void* {
    void* p = base + off;
    off = (off + bytes + 255) & ~(size_t)255;
    return p;
  };
  const size_t NF = (size_t)N_NODES * HID;
  float*   A    = (float*)  alloc(NF * 4);          // agg buf / out_feat (reused)
  bf16_t*  actA = (bf16_t*) alloc(NF * 2);          // activation bf16
  bf16_t*  actB = (bf16_t*) alloc(NF * 2);          // agg bf16
  float*   H1   = (float*)  alloc(NF * 4);          // h1, later ht
  float*   H2   = (float*)  alloc(NF * 4);          // h2
  bf16_t*  W1rt = (bf16_t*) alloc(HID * HID * 2);
  bf16_t*  W1nt = (bf16_t*) alloc(HID * HID * 2);
  bf16_t*  W2rt = (bf16_t*) alloc(HID * HID * 2);
  bf16_t*  W2nt = (bf16_t*) alloc(HID * HID * 2);
  bf16_t*  Wat  = (bf16_t*) alloc(HID * HID * 2);
  float*   sS   = (float*)  alloc(N_NODES * 4);
  float*   sD   = (float*)  alloc(N_NODES * 4);
  unsigned* mkey= (unsigned*)alloc(N_NODES * 4);
  float*   denom= (float*)  alloc(N_NODES * 4);
  float*   eV   = (float*)  alloc((size_t)N_EDGES * 4);
  float*   eE   = (float*)  alloc((size_t)N_EDGES * 4);
  float*   HT   = H1;                               // alias: H1 free when ht computed
  float*   OF   = A;                                // alias: agg free when out_feat built

  const int TB = 256;
  const long long nf4 = (long long)NF / 4;
  const int g_nf4   = (int)((nf4 + TB - 1) / TB);
  const int g_node4 = (N_NODES / 4 + TB - 1) / TB;
  const int g_scat  = (int)(((long long)N_EDGES * 32 + TB - 1) / TB);
  const int g_edge  = (N_EDGES + TB - 1) / TB;
  const int g_gemm  = N_NODES / 16;                 // one block per 16-row M tile
  const int g_wavesN= (N_NODES * 32 + TB - 1) / TB; // wave-per-node kernels

  // weight prep (bf16, transposed)
  k_wtrans<<<64, TB, 0, stream>>>(W1r, W1rt);
  k_wtrans<<<64, TB, 0, stream>>>(W1n, W1nt);
  k_wtrans<<<64, TB, 0, stream>>>(W2r, W2rt);
  k_wtrans<<<64, TB, 0, stream>>>(W2n, W2nt);
  k_wtrans<<<64, TB, 0, stream>>>(Wa,  Wat);

  // --- layer 1: h1 = relu(x@W1r + agg(x)@W1n + b1) ---
  k_zero4<<<g_nf4, TB, 0, stream>>>(A, nf4);
  k_scatter_add<<<g_scat, TB, 0, stream>>>(x, src, dst, A, N_EDGES);
  k_cvt_bf16<<<g_nf4, TB, 0, stream>>>(x, actA, nf4);
  k_cvt_bf16<<<g_nf4, TB, 0, stream>>>(A, actB, nf4);
  k_gemm_bf16<<<g_gemm, TB, 0, stream>>>(actA, actB, W1rt, W1nt, b1, H1, 1, 1);

  // --- layer 2: h2 = relu(h1@W2r + agg(h1)@W2n + b2) ---
  k_zero4<<<g_nf4, TB, 0, stream>>>(A, nf4);
  k_scatter_add<<<g_scat, TB, 0, stream>>>(H1, src, dst, A, N_EDGES);
  k_cvt_bf16<<<g_nf4, TB, 0, stream>>>(H1, actA, nf4);
  k_cvt_bf16<<<g_nf4, TB, 0, stream>>>(A, actB, nf4);
  k_gemm_bf16<<<g_gemm, TB, 0, stream>>>(actA, actB, W2rt, W2nt, b2, H2, 1, 1);

  // --- ht = h2 @ Wa (no bias / relu); HT aliases H1 (free by now) ---
  k_cvt_bf16<<<g_nf4, TB, 0, stream>>>(H2, actA, nf4);
  k_gemm_bf16<<<g_gemm, TB, 0, stream>>>(actA, actA, Wat, Wat, nullptr, HT, 0, 0);

  // --- attention scores + segment softmax ---
  k_attn_scores<<<g_wavesN, TB, 0, stream>>>(HT, a_src, a_dst, sS, sD, N_NODES);
  k_zero4<<<g_node4, TB, 0, stream>>>((float*)mkey, N_NODES / 4); // key 0 == -inf
  k_zero4<<<g_node4, TB, 0, stream>>>(denom, N_NODES / 4);
  k_edge_logits<<<g_edge, TB, 0, stream>>>(sS, sD, src, dst, eV, mkey, N_EDGES);
  k_edge_exp<<<g_edge, TB, 0, stream>>>(eV, mkey, dst, eE, denom, N_EDGES);

  // --- out_feat = segment_sum(alpha * ht[src]); OF aliases A (re-zeroed) ---
  k_zero4<<<g_nf4, TB, 0, stream>>>(OF, nf4);
  k_weighted_scatter<<<g_scat, TB, 0, stream>>>(eE, denom, src, dst, HT, OF,
                                                N_EDGES);

  // --- final FC -> [N, 2] ---
  k_final_fc<<<g_wavesN, TB, 0, stream>>>(OF, Wfc, bfc, out, N_NODES);
}
